// Attention_57251914056149
// MI455X (gfx1250) — compile-verified, matrix operand order checked by reading, not run on hardware
//
#include <hip/hip_runtime.h>
#include <hip/hip_bf16.h>
#include <math.h>

// ---------------------------------------------------------------------------
// Fused attention for B=16, T=1024, N_EMBD=1024 on gfx1250 (MI455X).
// v2: weights pre-converted to bf16 (kills in-loop v_cvt + WMMA WAR-hazard
//     nops), wider wave tiles (one A-fragment feeds 4-8 WMMAs), grouped
//     j-tiles / e-tiles in the attention kernel for operand reuse.
// Pipeline:
//   k0: convert Wq/Wk/Wv fp32 -> bf16 workspace
//   k1: QKV projection (X @ W^T + b) -> bf16 Q,K row-major; V transposed
//   k2: per (batch, 16-row q tile): S = Q K^T in LDS, mask, softmax,
//       O = P V with P in LDS as bf16.
// ---------------------------------------------------------------------------

#define NE   1024            // embedding dim == seq len
#define NB   16              // batch
#define NTOK (NB * NE)       // 16384 token rows

typedef __attribute__((ext_vector_type(16))) __bf16 v16bf;
typedef __attribute__((ext_vector_type(8)))  __bf16 v8bf;
typedef __attribute__((ext_vector_type(8)))  float  v8f;
typedef __attribute__((ext_vector_type(4)))  float  v4f;

// A/B fragment K mapping (16-bit, 16x32): lane half h = lane>>4 reads two
// contiguous 8-element runs: [k0+8h .. +7] and [k0+16+8h .. +7].

__device__ inline v16bf frag_from_f32(const float* __restrict__ p) {
  v4f a0 = *(const v4f*)(p);
  v4f a1 = *(const v4f*)(p + 4);
  v4f a2 = *(const v4f*)(p + 16);
  v4f a3 = *(const v4f*)(p + 20);
  v16bf f;
#pragma unroll
  for (int i = 0; i < 4; ++i) {
    f[i]      = (__bf16)a0[i];
    f[4 + i]  = (__bf16)a1[i];
    f[8 + i]  = (__bf16)a2[i];
    f[12 + i] = (__bf16)a3[i];
  }
  return f;
}

__device__ inline v16bf frag_from_bf16(const __bf16* p) {
  v8bf lo = *(const v8bf*)(p);
  v8bf hi = *(const v8bf*)(p + 16);
  v16bf f;
#pragma unroll
  for (int i = 0; i < 8; ++i) { f[i] = lo[i]; f[8 + i] = hi[i]; }
  return f;
}

__device__ inline v8f wmma_bf16(v16bf a, v16bf b, v8f c) {
  return __builtin_amdgcn_wmma_f32_16x16x32_bf16(
      /*neg_a=*/false, a, /*neg_b=*/false, b,
      /*c_mod=*/(short)0, c, /*reuse_a=*/false, /*reuse_b=*/false);
}

// ---------------------------------------------------------------------------
// Kernel 0: convert the three weight matrices fp32 -> bf16 (once).
// grid = (NE*NE/2048, 3), block = 256; 8 elements per thread.
// ---------------------------------------------------------------------------
__global__ __launch_bounds__(256) void cvt_w_kernel(
    const float* __restrict__ Wq, const float* __restrict__ Wk,
    const float* __restrict__ Wv, __bf16* __restrict__ out)
{
  const int z = blockIdx.y;
  const float* src = (z == 0) ? Wq : (z == 1) ? Wk : Wv;
  __bf16* dst = out + (size_t)z * NE * NE;
  const size_t i = ((size_t)blockIdx.x * 256 + threadIdx.x) * 8;
  v4f a = *(const v4f*)(src + i);
  v4f b = *(const v4f*)(src + i + 4);
  v8bf o;
#pragma unroll
  for (int j = 0; j < 4; ++j) { o[j] = (__bf16)a[j]; o[4 + j] = (__bf16)b[j]; }
  *(v8bf*)(dst + i) = o;
}

// ---------------------------------------------------------------------------
// Kernel 1: QKV projection.  grid = (NTOK/16, 2, 3), block = 128 (4 waves).
// Wave computes 16(M) x 128(N): one A-fragment feeds 8 WMMAs.
//   C[m,n] = sum_k X[m,k] * Wbf[n,k] + b[n]
// ---------------------------------------------------------------------------
__global__ __launch_bounds__(128) void qkv_proj_kernel(
    const float* __restrict__ Xq, const float* __restrict__ Xk,
    const float* __restrict__ Xv,
    const __bf16* __restrict__ Wb,
    const float* __restrict__ bq, const float* __restrict__ bk,
    const float* __restrict__ bv,
    __bf16* __restrict__ Qo, __bf16* __restrict__ Ko,
    __bf16* __restrict__ Vt)
{
  const int z = blockIdx.z;
  const float*  X    = (z == 0) ? Xq : (z == 1) ? Xk : Xv;
  const float*  bias = (z == 0) ? bq : (z == 1) ? bk : bv;
  const __bf16* W    = Wb + (size_t)z * NE * NE;

  const int wave = threadIdx.x >> 5;
  const int lane = threadIdx.x & 31;
  const int half = lane >> 4;
  const int lr   = lane & 15;

  const int m0 = blockIdx.x * 16;                   // token-row tile
  const int n0 = (blockIdx.y * 4 + wave) * 128;     // output-feature base

  v8f acc[8];
#pragma unroll
  for (int t = 0; t < 8; ++t)
#pragma unroll
    for (int r = 0; r < 8; ++r) acc[t][r] = 0.0f;

  const float* arow = X + (size_t)(m0 + lr) * NE + 8 * half;

  for (int k0 = 0; k0 < NE; k0 += 32) {
    v16bf a = frag_from_f32(arow + k0);             // only conversion in loop
#pragma unroll
    for (int nt = 0; nt < 8; ++nt) {
      const __bf16* brow = W + (size_t)(n0 + nt * 16 + lr) * NE + 8 * half;
      v16bf b = frag_from_bf16(brow + k0);          // pure b128 load
      acc[nt] = wmma_bf16(a, b, acc[nt]);
    }
  }

  const int bidx = m0 >> 10;      // batch of this tile
  const int t0   = m0 & (NE - 1); // token within batch

#pragma unroll
  for (int nt = 0; nt < 8; ++nt) {
    const int n  = n0 + nt * 16 + lr;
    const float bb = bias[n];
    if (z < 2) {
      __bf16* out = (z == 0) ? Qo : Ko;   // row-major [NTOK][NE]
#pragma unroll
      for (int r = 0; r < 8; ++r)
        out[(size_t)(m0 + 8 * half + r) * NE + n] = (__bf16)(acc[nt][r] + bb);
    } else {
      // V stored transposed: Vt[b][e][t]; contiguous in t per lane.
      v8bf pk;
#pragma unroll
      for (int r = 0; r < 8; ++r) pk[r] = (__bf16)(acc[nt][r] + bb);
      *(v8bf*)(Vt + (size_t)bidx * NE * NE + (size_t)n * NE + t0 + 8 * half) = pk;
    }
  }
}

// ---------------------------------------------------------------------------
// Kernel 2: attention.  grid = NB*64 blocks, block = 64 threads (2 waves).
// Each block handles one (batch b, q-tile of 16 rows).
// ---------------------------------------------------------------------------
__global__ __launch_bounds__(64) void attn_kernel(
    const __bf16* __restrict__ Q, const __bf16* __restrict__ K,
    const __bf16* __restrict__ Vt, float* __restrict__ O,
    const int* __restrict__ maskp)
{
  __shared__ __align__(16) float  Sbuf[16 * NE];   // 64 KB scores
  __shared__ __align__(16) __bf16 Pbuf[16 * NE];   // 32 KB probabilities
  __shared__ float redm[16][4];
  __shared__ float reds[16][4];

  const int b  = blockIdx.x >> 6;
  const int qt = blockIdx.x & 63;
  const int q0 = qt * 16;

  const int tid  = threadIdx.x;
  const int wave = tid >> 5;
  const int lane = tid & 31;
  const int half = lane >> 4;
  const int lr   = lane & 15;

  const int causal = *maskp;
  const int jmax   = causal ? qt : 63;           // last k-tile (inclusive)
  const int L      = (jmax + 1) * 16;            // valid key length

  const __bf16* Qb  = Q  + (size_t)b * NE * NE;
  const __bf16* Kb  = K  + (size_t)b * NE * NE;
  const __bf16* Vtb = Vt + (size_t)b * NE * NE;
  float*        Ob  = O  + (size_t)b * NE * NE;

  // ---- Phase 1: S = (Q K^T)/32 + mask -> LDS ----------------------------
  // Wave handles j-tiles {wave, wave+2, ...}; grouped 4 at a time so one Q
  // A-fragment feeds 4 WMMAs.
  const __bf16* qrow = Qb + (size_t)(q0 + lr) * NE + 8 * half;
  const int ntiles = (jmax >= wave) ? (((jmax - wave) >> 1) + 1) : 0;
  int jb = wave;

  for (int grp = 0; grp < (ntiles >> 2); ++grp, jb += 8) {   // full groups
    v8f acc[4];
#pragma unroll
    for (int g = 0; g < 4; ++g)
#pragma unroll
      for (int r = 0; r < 8; ++r) acc[g][r] = 0.0f;
#pragma unroll 2
    for (int k0 = 0; k0 < NE; k0 += 32) {
      v16bf a = frag_from_bf16(qrow + k0);
#pragma unroll
      for (int g = 0; g < 4; ++g) {
        const __bf16* krow =
            Kb + (size_t)((jb + 2 * g) * 16 + lr) * NE + 8 * half;
        v16bf bf = frag_from_bf16(krow + k0);
        acc[g] = wmma_bf16(a, bf, acc[g]);
      }
    }
#pragma unroll
    for (int g = 0; g < 4; ++g) {
      const int kcol = (jb + 2 * g) * 16 + lr;
#pragma unroll
      for (int r = 0; r < 8; ++r) {
        const int qg = q0 + 8 * half + r;
        float s = acc[g][r] * 0.03125f;
        if (causal && kcol > qg) s = -INFINITY;
        Sbuf[(8 * half + r) * NE + kcol] = s;
      }
    }
  }
  for (int rem = 0; rem < (ntiles & 3); ++rem, jb += 2) {    // remainder
    v8f acc;
#pragma unroll
    for (int r = 0; r < 8; ++r) acc[r] = 0.0f;
    const __bf16* krow = Kb + (size_t)(jb * 16 + lr) * NE + 8 * half;
#pragma unroll 4
    for (int k0 = 0; k0 < NE; k0 += 32) {
      v16bf a  = frag_from_bf16(qrow + k0);
      v16bf bf = frag_from_bf16(krow + k0);
      acc = wmma_bf16(a, bf, acc);
    }
    const int kcol = jb * 16 + lr;
#pragma unroll
    for (int r = 0; r < 8; ++r) {
      const int qg = q0 + 8 * half + r;
      float s = acc[r] * 0.03125f;
      if (causal && kcol > qg) s = -INFINITY;
      Sbuf[(8 * half + r) * NE + kcol] = s;
    }
  }
  __syncthreads();

  // ---- Phase 2: softmax over the L valid keys of each of 16 rows --------
  {
    const int row = tid & 15;
    const int seg = tid >> 4;                    // 4 segments per row
    const int seglen = L >> 2;                   // L multiple of 16
    const int kbeg = seg * seglen, kend = kbeg + seglen;

    float m = -INFINITY;
    for (int k = kbeg; k < kend; ++k) m = fmaxf(m, Sbuf[row * NE + k]);
    redm[row][seg] = m;
    __syncthreads();
    const float M = fmaxf(fmaxf(redm[row][0], redm[row][1]),
                          fmaxf(redm[row][2], redm[row][3]));
    float sum = 0.0f;
    for (int k = kbeg; k < kend; ++k) {
      float e = __expf(Sbuf[row * NE + k] - M);
      Sbuf[row * NE + k] = e;
      sum += e;
    }
    reds[row][seg] = sum;
    __syncthreads();
    const float inv =
        1.0f / (reds[row][0] + reds[row][1] + reds[row][2] + reds[row][3]);
    for (int k = kbeg; k < kend; ++k)
      Pbuf[row * NE + k] = (__bf16)(Sbuf[row * NE + k] * inv);
    // zero-pad P to a 32-multiple of K for the WMMA chunking below
    if (tid < 16 && L < NE) {
      for (int k = L; k < L + 16; ++k) Pbuf[tid * NE + k] = (__bf16)0.0f;
    }
  }
  __syncthreads();

  // ---- Phase 3: O = P V.  e-tiles grouped 4 at a time: one P A-fragment
  // (2x ds_load_b128) feeds 4 WMMAs.  32 tiles/wave = 8 exact groups. ------
  const int nch = (L + 31) / 32;                 // K chunks of 32
  const __bf16* prow = &Pbuf[lr * NE + 8 * half];
  for (int eg = wave; eg < 16; eg += 2) {
    const int e0 = eg * 64;                      // 4 tiles of 16 columns
    v8f acc[4];
#pragma unroll
    for (int t = 0; t < 4; ++t)
#pragma unroll
      for (int r = 0; r < 8; ++r) acc[t][r] = 0.0f;

    for (int c = 0; c < nch; ++c) {
      const int k0 = c * 32;
      v16bf a = frag_from_bf16(prow + k0);       // LDS
#pragma unroll
      for (int t = 0; t < 4; ++t) {
        const __bf16* vrow =
            Vtb + (size_t)(e0 + t * 16 + lr) * NE + 8 * half;
        v16bf bf = frag_from_bf16(vrow + k0);
        acc[t] = wmma_bf16(a, bf, acc[t]);
      }
    }
#pragma unroll
    for (int t = 0; t < 4; ++t)
#pragma unroll
      for (int r = 0; r < 8; ++r)
        Ob[(size_t)(q0 + 8 * half + r) * NE + e0 + t * 16 + lr] = acc[t][r];
  }
}

// ---------------------------------------------------------------------------
extern "C" void kernel_launch(void* const* d_in, const int* in_sizes, int n_in,
                              void* d_out, int out_size, void* d_ws,
                              size_t ws_size, hipStream_t stream) {
  (void)in_sizes; (void)n_in; (void)out_size; (void)ws_size;

  const float* xq = (const float*)d_in[0];
  const float* xk = (const float*)d_in[1];
  const float* xv = (const float*)d_in[2];
  const float* Wq = (const float*)d_in[3];
  const float* bq = (const float*)d_in[4];
  const float* Wk = (const float*)d_in[5];
  const float* bk = (const float*)d_in[6];
  const float* Wv = (const float*)d_in[7];
  const float* bv = (const float*)d_in[8];
  const int* maskp = (const int*)d_in[9];

  // Workspace layout (bytes): [0, 6M) Wbf | 8M Qbf | 40M Kbf | 72M Vt
  char* ws = (char*)d_ws;
  __bf16* Wbf = reinterpret_cast<__bf16*>(ws);
  __bf16* Qb  = reinterpret_cast<__bf16*>(ws + (size_t)8  * 1024 * 1024);
  __bf16* Kb  = reinterpret_cast<__bf16*>(ws + (size_t)40 * 1024 * 1024);
  __bf16* Vt  = reinterpret_cast<__bf16*>(ws + (size_t)72 * 1024 * 1024);

  cvt_w_kernel<<<dim3(NE * NE / 2048, 3), 256, 0, stream>>>(Wq, Wk, Wv, Wbf);

  qkv_proj_kernel<<<dim3(NTOK / 16, 2, 3), 128, 0, stream>>>(
      xq, xk, xv, Wbf, bq, bk, bv, Qb, Kb, Vt);

  attn_kernel<<<dim3(NB * 64), 64, 0, stream>>>(Qb, Kb, Vt, (float*)d_out,
                                                maskp);
}
